// HGT_Convs_50070728737143
// MI455X (gfx1250) — matrix operand order repaired and checked by reading, not compiled, not run on hardware
//
#include <hip/hip_runtime.h>
#include <hip/hip_bf16.h>
#include <math.h>

#define NGENE  20000
#define NPROT  20000
#define NEDGE  320000
#define DIM    128
#define NLAYER 2
#define NTYPE  2
#define NREL   3
#define DD     (DIM * DIM)

typedef __attribute__((ext_vector_type(16))) __bf16 v16bf;
typedef __attribute__((ext_vector_type(8)))  float  v8f;

// ---------------------------------------------------------------------------
// helpers
// ---------------------------------------------------------------------------
__device__ __forceinline__ unsigned int pack_bf16(float lo, float hi) {
  float2 t; t.x = lo; t.y = hi;
  union { __hip_bfloat162 h2; unsigned int u; } cvt;
  cvt.h2 = __float22bfloat162_rn(t);     // packed f32 -> bf16 convert
  return cvt.u;
}

// order-preserving float<->uint encoding for atomicMax-based segment max
__device__ __forceinline__ unsigned int enc_f32(float x) {
  unsigned int u = __float_as_uint(x);
  return (u >> 31) ? ~u : (u | 0x80000000u);
}
__device__ __forceinline__ float dec_f32(unsigned int e) {
  unsigned int u = (e >> 31) ? (e ^ 0x80000000u) : ~e;
  return __uint_as_float(u);
}

union FragAB { v16bf v; uint4 q[2]; };

// ---------------------------------------------------------------------------
// Generic GEMM: C[M x 128] = A[M x 128] @ W[128 x 128] (+ bias[128])
// A, W are f32 in global; converted to bf16 while staging into LDS.
// Block: 256 threads (8 waves). Tile: 128 rows. Each wave: 16 rows x 8 WMMA
// accumulators, K swept in 4 steps of 32 via v_wmma_f32_16x16x32_bf16.
// W is staged straight into the WMMA B-fragment layout (two contiguous
// ds_load_b128 per fragment). Per k-step all 8 B fragments are batch-loaded
// and fenced with a sched_barrier so each WMMA only waits for its own
// fragment (incremental s_wait_dscnt) instead of stalling on dscnt 0.
// ---------------------------------------------------------------------------
__global__ __launch_bounds__(256)
void gemm_bf16_128(const float* __restrict__ A, int M,
                   const float* __restrict__ W,
                   const float* __restrict__ bias,
                   float* __restrict__ C)
{
  __shared__ unsigned short sA[128][136];      // row stride 272B = 16*17 (16B aligned)
  __shared__ unsigned int   sWf[4][8][32][8];  // [k-step][n-tile][lane][8 packed dwords]

  const int tid = threadIdx.x;
  const int m0  = blockIdx.x * 128;

  // ---- stage A tile as bf16 (packed 64-bit LDS stores, zero-padded past M) ----
  {
    int row = tid >> 1;
    int cb  = (tid & 1) * 64;
    int gm  = m0 + row;
    const float4* arow = (const float4*)(A + (size_t)gm * DIM + cb);
#pragma unroll
    for (int j = 0; j < 16; ++j) {
      float4 fa = (gm < M) ? arow[j] : make_float4(0.f, 0.f, 0.f, 0.f);
      uint2 p;
      p.x = pack_bf16(fa.x, fa.y);
      p.y = pack_bf16(fa.z, fa.w);
      *(uint2*)&sA[row][cb + j * 4] = p;
    }
  }
  // ---- stage W directly in B-fragment layout ----
  // fragment (k0s, nt): lane covers n = nt*16 + lane%16, K-halves by lane/16;
  // dword j packs B[kb+2j][n], B[kb+2j+1][n].
  {
#pragma unroll
    for (int i = 0; i < 4; ++i) {
      int tr   = tid * 4 + i;            // 0..1023 = (k0s*8 + nt)*32 + lane
      int k0s  = tr >> 8;
      int nt   = (tr >> 5) & 7;
      int lane = tr & 31;
      int n    = nt * 16 + (lane & 15);
      int kb   = k0s * 32 + ((lane >> 4) << 4);
#pragma unroll
      for (int j = 0; j < 8; ++j) {
        float w0 = W[(size_t)(kb + 2 * j)     * DIM + n];
        float w1 = W[(size_t)(kb + 2 * j + 1) * DIM + n];
        sWf[k0s][nt][lane][j] = pack_bf16(w0, w1);
      }
    }
  }
  __syncthreads();

  const int wave  = tid >> 5;
  const int lane  = tid & 31;
  const int mw    = wave * 16;     // wave's 16-row slab within the tile
  const int lhalf = lane >> 4;     // 0: lanes 0-15, 1: lanes 16-31
  const int l16   = lane & 15;

  v8f vzero = {};
  v8f acc[8];
#pragma unroll
  for (int i = 0; i < 8; ++i) acc[i] = vzero;

  const uint4* ap = (const uint4*)&sA[mw + l16][0];

  FragAB afr[2];
  afr[0].q[0] = ap[2 * lhalf];             // k-step 0: kb/8 = 2*lhalf
  afr[0].q[1] = ap[2 * lhalf + 1];

#pragma unroll
  for (int k0s = 0; k0s < 4; ++k0s) {
    const int pa = k0s & 1;
    // batch-load all 8 B fragments for this k-step (16 x ds_load_b128)
    FragAB b[8];
#pragma unroll
    for (int nt = 0; nt < 8; ++nt) {
      const uint4* bq = (const uint4*)&sWf[k0s][nt][lane][0];
      b[nt].q[0] = bq[0];
      b[nt].q[1] = bq[1];
    }
    if (k0s < 3) {                         // queue next A behind the B batch
      afr[pa ^ 1].q[0] = ap[(k0s + 1) * 4 + 2 * lhalf];
      afr[pa ^ 1].q[1] = ap[(k0s + 1) * 4 + 2 * lhalf + 1];
    }
    __builtin_amdgcn_sched_barrier(0);     // loads stay ahead of the WMMA block
#pragma unroll
    for (int nt = 0; nt < 8; ++nt) {
      acc[nt] = __builtin_amdgcn_wmma_f32_16x16x32_bf16(
          false, afr[pa].v, false, b[nt].v, (short)0, acc[nt], false, false);
    }
  }

  // epilogue: D layout -> VGPR r is row (r + 8*lhalf), column l16
  const int mb = m0 + mw + lhalf * 8;
  if (m0 + 128 <= M) {                     // full tile: unguarded stores
#pragma unroll
    for (int nt = 0; nt < 8; ++nt) {
      int   n  = nt * 16 + l16;
      float bv = bias ? bias[n] : 0.0f;
#pragma unroll
      for (int r = 0; r < 8; ++r)
        C[(size_t)(mb + r) * DIM + n] = acc[nt][r] + bv;
    }
  } else {
#pragma unroll
    for (int nt = 0; nt < 8; ++nt) {
      int   n  = nt * 16 + l16;
      float bv = bias ? bias[n] : 0.0f;
#pragma unroll
      for (int r = 0; r < 8; ++r) {
        int gm = mb + r;
        if (gm < M) C[(size_t)gm * DIM + n] = acc[nt][r] + bv;
      }
    }
  }
}

// ---------------------------------------------------------------------------
// Edge phase kernels (wave32 = one edge per wave, lanes cover the 128 feats)
// ---------------------------------------------------------------------------
__global__ __launch_bounds__(256)
void edge_score(const float* __restrict__ q, const float* __restrict__ kr,
                const int* __restrict__ src, const int* __restrict__ dst,
                const float* __restrict__ prel, float* __restrict__ sout,
                unsigned int* __restrict__ mx, int nE)
{
  int e    = (int)((blockIdx.x * blockDim.x + threadIdx.x) >> 5);
  int lane = threadIdx.x & 31;
  if (e >= nE) return;
  int si = src[e], di = dst[e];
  float4 qa = ((const float4*)(q  + (size_t)di * DIM))[lane];
  float4 ka = ((const float4*)(kr + (size_t)si * DIM))[lane];
  float d = qa.x * ka.x + qa.y * ka.y + qa.z * ka.z + qa.w * ka.w;
#pragma unroll
  for (int off = 16; off > 0; off >>= 1) d += __shfl_xor(d, off, 32);
  if (lane == 0) {
    float sc = d * prel[0] * 0.08838834764831845f;   // * p_rel * 1/sqrt(128)
    sout[e] = sc;
    atomicMax(mx + di, enc_f32(sc));
  }
}

__global__ __launch_bounds__(256)
void edge_exp(const int* __restrict__ dst, const unsigned int* __restrict__ mx,
              float* __restrict__ s, float* __restrict__ den, int nE)
{
  int e = blockIdx.x * blockDim.x + threadIdx.x;
  if (e >= nE) return;
  int di  = dst[e];
  float m = dec_f32(mx[di]);
  if (!isfinite(m)) m = 0.0f;
  float ex = expf(s[e] - m);
  s[e] = ex;                       // scores buffer now holds exp(s - m)
  atomicAdd(den + di, ex);
}

__global__ __launch_bounds__(256)
void edge_scatter(const float* __restrict__ vr, const int* __restrict__ src,
                  const int* __restrict__ dst, const float* __restrict__ ex,
                  const float* __restrict__ den, float* __restrict__ agg, int nE)
{
  int e    = (int)((blockIdx.x * blockDim.x + threadIdx.x) >> 5);
  int lane = threadIdx.x & 31;
  if (e >= nE) return;
  int si = src[e], di = dst[e];
  float a  = ex[e] / den[di];
  float4 v = ((const float4*)(vr + (size_t)si * DIM))[lane];
  float* ap = agg + (size_t)di * DIM + lane * 4;
  atomicAdd(ap + 0, a * v.x);
  atomicAdd(ap + 1, a * v.y);
  atomicAdd(ap + 2, a * v.z);
  atomicAdd(ap + 3, a * v.w);
}

// ---------------------------------------------------------------------------
// Elementwise kernels
// ---------------------------------------------------------------------------
__global__ __launch_bounds__(256)
void gelu_inplace(float* __restrict__ x, int n)
{
  int i = blockIdx.x * blockDim.x + threadIdx.x;
  if (i < n) {
    float v = x[i];
    x[i] = 0.5f * v * (1.0f + erff(v * 0.7071067811865475f));  // exact gelu
  }
}

__global__ __launch_bounds__(256)
void skip_relu(const float* __restrict__ o, const float* __restrict__ skip,
               float* __restrict__ cur, int n)
{
  int i = blockIdx.x * blockDim.x + threadIdx.x;
  if (i < n) {
    float g = 1.0f / (1.0f + expf(-skip[0]));
    float r = g * o[i] + (1.0f - g) * cur[i];
    cur[i] = r > 0.0f ? r : 0.0f;
  }
}

// ---------------------------------------------------------------------------
// Orchestration
// ---------------------------------------------------------------------------
extern "C" void kernel_launch(void* const* d_in, const int* in_sizes, int n_in,
                              void* d_out, int out_size, void* d_ws, size_t ws_size,
                              hipStream_t stream)
{
  (void)in_sizes; (void)n_in; (void)out_size; (void)ws_size;

  const float* x_gene = (const float*)d_in[0];
  const float* x_prot = (const float*)d_in[1];
  const int*   ei[3]  = {(const int*)d_in[2], (const int*)d_in[3], (const int*)d_in[4]};
  const float* Wk     = (const float*)d_in[5];
  const float* bk     = (const float*)d_in[6];
  const float* Wq     = (const float*)d_in[7];
  const float* bq     = (const float*)d_in[8];
  const float* Wv     = (const float*)d_in[9];
  const float* bv     = (const float*)d_in[10];
  const float* Wo     = (const float*)d_in[11];
  const float* bo     = (const float*)d_in[12];
  const float* skip   = (const float*)d_in[13];
  const float* a_rel  = (const float*)d_in[14];
  const float* m_rel  = (const float*)d_in[15];
  const float* p_rel  = (const float*)d_in[16];

  const size_t ND = (size_t)NGENE * DIM;   // gene and protein counts are equal

  // workspace layout (floats)
  float* ws = (float*)d_ws;
  float* cur[2] = { ws,           ws + ND };
  float* q[2]   = { ws + 2 * ND,  ws + 3 * ND };
  float* kr[3], *vr[3];
  for (int r = 0; r < NREL; ++r) { kr[r] = ws + (4 + r) * ND; vr[r] = ws + (7 + r) * ND; }
  float* agg[2] = { ws + 10 * ND, ws + 11 * ND };
  float* o[2]   = { q[0], q[1] };          // reuse: q is dead once scores are done
  float* sc[3];
  sc[0] = ws + 12 * ND; sc[1] = sc[0] + NEDGE; sc[2] = sc[1] + NEDGE;
  float* den[2] = { sc[2] + NEDGE, sc[2] + NEDGE + NGENE };
  unsigned int* mx[2];
  mx[0] = (unsigned int*)(den[1] + NPROT);
  mx[1] = mx[0] + NGENE;
  float* fW = (float*)(mx[1] + NPROT);     // 6 fused 128x128 matrices
  float* WkA[3], *WvM[3], *bkA[3], *bvM[3];
  float* fb = fW + 6 * DD;                 // 6 fused bias vectors
  for (int r = 0; r < NREL; ++r) {
    WkA[r] = fW + r * DD;       WvM[r] = fW + (3 + r) * DD;
    bkA[r] = fb + r * DIM;      bvM[r] = fb + (3 + r) * DIM;
  }

  const int relsrc[NREL] = {0, 1, 0};      // REL = [(0,1),(1,0),(0,0)]
  const int reldst[NREL] = {1, 0, 0};
  const int gN   = (NGENE + 127) / 128;    // 157 GEMM tiles
  const int eWav = NEDGE / 8;              // 8 edges (waves) per 256-thread block
  const int eThr = (NEDGE + 255) / 256;
  const int nThr = ((int)ND + 255) / 256;

  hipMemcpyAsync(cur[0], x_gene, ND * sizeof(float), hipMemcpyDeviceToDevice, stream);
  hipMemcpyAsync(cur[1], x_prot, ND * sizeof(float), hipMemcpyDeviceToDevice, stream);

  for (int l = 0; l < NLAYER; ++l) {
    // fuse relation transforms into projection weights:
    // WkA[r] = Wk[l,st] @ a_rel[l,r]; bkA[r] = bk[l,st] @ a_rel[l,r] (same for V/m_rel)
    for (int r = 0; r < NREL; ++r) {
      int st = relsrc[r];
      const float* Ar = a_rel + ((size_t)l * NREL + r) * DD;
      const float* Mr = m_rel + ((size_t)l * NREL + r) * DD;
      gemm_bf16_128<<<1, 256, 0, stream>>>(Wk + ((size_t)l * NTYPE + st) * DD, DIM, Ar, nullptr, WkA[r]);
      gemm_bf16_128<<<1, 256, 0, stream>>>(bk + ((size_t)l * NTYPE + st) * DIM, 1,  Ar, nullptr, bkA[r]);
      gemm_bf16_128<<<1, 256, 0, stream>>>(Wv + ((size_t)l * NTYPE + st) * DD, DIM, Mr, nullptr, WvM[r]);
      gemm_bf16_128<<<1, 256, 0, stream>>>(bv + ((size_t)l * NTYPE + st) * DIM, 1,  Mr, nullptr, bvM[r]);
    }
    // Q projections per node type
    for (int t = 0; t < NTYPE; ++t)
      gemm_bf16_128<<<gN, 256, 0, stream>>>(cur[t], NGENE,
          Wq + ((size_t)l * NTYPE + t) * DD, bq + ((size_t)l * NTYPE + t) * DIM, q[t]);
    // fused K/V per relation
    for (int r = 0; r < NREL; ++r) {
      gemm_bf16_128<<<gN, 256, 0, stream>>>(cur[relsrc[r]], NGENE, WkA[r], bkA[r], kr[r]);
      gemm_bf16_128<<<gN, 256, 0, stream>>>(cur[relsrc[r]], NGENE, WvM[r], bvM[r], vr[r]);
    }
    // segment softmax statistics
    hipMemsetAsync(mx[0], 0, NGENE * sizeof(unsigned int), stream);
    hipMemsetAsync(mx[1], 0, NPROT * sizeof(unsigned int), stream);
    hipMemsetAsync(den[0], 0, NGENE * sizeof(float), stream);
    hipMemsetAsync(den[1], 0, NPROT * sizeof(float), stream);
    for (int r = 0; r < NREL; ++r)
      edge_score<<<eWav, 256, 0, stream>>>(q[reldst[r]], kr[r], ei[r], ei[r] + NEDGE,
                                           p_rel + l * NREL + r, sc[r], mx[reldst[r]], NEDGE);
    for (int r = 0; r < NREL; ++r)
      edge_exp<<<eThr, 256, 0, stream>>>(ei[r] + NEDGE, mx[reldst[r]], sc[r], den[reldst[r]], NEDGE);
    // weighted aggregation
    hipMemsetAsync(agg[0], 0, ND * sizeof(float), stream);
    hipMemsetAsync(agg[1], 0, ND * sizeof(float), stream);
    for (int r = 0; r < NREL; ++r)
      edge_scatter<<<eWav, 256, 0, stream>>>(vr[r], ei[r], ei[r] + NEDGE, sc[r],
                                             den[reldst[r]], agg[reldst[r]], NEDGE);
    // gelu -> output projection -> gated skip + relu (writes cur in place)
    for (int t = 0; t < NTYPE; ++t) {
      gelu_inplace<<<nThr, 256, 0, stream>>>(agg[t], (int)ND);
      gemm_bf16_128<<<gN, 256, 0, stream>>>(agg[t], NGENE,
          Wo + ((size_t)l * NTYPE + t) * DD, bo + ((size_t)l * NTYPE + t) * DIM, o[t]);
      skip_relu<<<nThr, 256, 0, stream>>>(o[t], skip + l * NTYPE + t, cur[t], (int)ND);
    }
  }

  hipMemcpyAsync(d_out, cur[0], ND * sizeof(float), hipMemcpyDeviceToDevice, stream);
  hipMemcpyAsync((float*)d_out + ND, cur[1], ND * sizeof(float), hipMemcpyDeviceToDevice, stream);
}
